// Attention_50122268344914
// MI455X (gfx1250) — matrix operand order repaired and checked by reading, not compiled, not run on hardware
//
#include <hip/hip_runtime.h>
#include <hip/hip_bf16.h>

#define DEV __device__ __forceinline__

typedef __attribute__((ext_vector_type(16))) __bf16 bf16x16;
typedef __attribute__((ext_vector_type(8)))  float  f32x8;

// Problem sizes (compile-time)
constexpr int BB = 4, NN = 2048, DD = 1024, HH = 16, HD = 64;
constexpr int M    = BB * NN;   // 8192 rows
constexpr int NOUT = 3 * DD;    // 3072 qkv cols

// ---- helpers ----------------------------------------------------------------

DEV unsigned short f2bf(float f) {              // round-to-nearest-even f32->bf16
  union { float f; unsigned u; } x; x.f = f;
  unsigned r = x.u + 0x7FFFu + ((x.u >> 16) & 1u);
  return (unsigned short)(r >> 16);
}

DEV f32x8 wmma_bf16(bf16x16 a, bf16x16 b, f32x8 c) {
  return __builtin_amdgcn_wmma_f32_16x16x32_bf16(false, a, false, b, (short)0, c,
                                                 false, false);
}

union fragU { bf16x16 v; uint4 u[2]; };

// A-matrix 16x32 bf16 fragment (row-major source, ld in elements).
// Lane holds row (lane&15); K contiguous in [half*16, half*16+16)
DEV bf16x16 load_A_frag(const unsigned short* base, int row, int ld, int kbase, int half) {
  fragU r;
  r.u[0] = *(const uint4*)(base + row * ld + kbase + half * 16);
  r.u[1] = *(const uint4*)(base + row * ld + kbase + half * 16 + 8);
  return r.v;
}

// B-matrix 32x16 bf16 fragment from contraction-contiguous storage [n][k].
// Lane holds col (lane&15); K contiguous in [half*16, half*16+16)
DEV bf16x16 load_B_frag(const unsigned short* base, int ncol, int ld, int kbase, int half) {
  const uint4* p = (const uint4*)(base + ncol * ld + kbase + half * 16);
  fragU r; r.u[0] = p[0]; r.u[1] = p[1];
  return r.v;
}

// CDNA5 async global->LDS copy (ASYNCcnt-tracked, no VGPR round trip).
DEV unsigned lds_off(const void* p) {           // low 32 bits of LDS flat addr = LDS byte offset
  return (unsigned)(size_t)p;
}
DEV void async_ld_b128(unsigned loff, const void* g) {
  asm volatile("global_load_async_to_lds_b128 %0, %1, off"
               :: "v"(loff), "v"((unsigned long long)g) : "memory");
}
DEV void wait_async0() { asm volatile("s_wait_asynccnt 0x0" ::: "memory"); }

// ---- prep kernels -----------------------------------------------------------

__global__ void k_convert(const float* __restrict__ src, unsigned short* __restrict__ dst,
                          int n8) {
  int i = blockIdx.x * blockDim.x + threadIdx.x;
  if (i >= n8) return;
  const float4* s = (const float4*)src + (size_t)i * 2;
  float4 a = s[0], b = s[1];
  union { unsigned short h[8]; uint4 u; } o;
  o.h[0] = f2bf(a.x); o.h[1] = f2bf(a.y); o.h[2] = f2bf(a.z); o.h[3] = f2bf(a.w);
  o.h[4] = f2bf(b.x); o.h[5] = f2bf(b.y); o.h[6] = f2bf(b.z); o.h[7] = f2bf(b.w);
  *(uint4*)(dst + (size_t)i * 8) = o.u;
}

// src [rows][cols] f32  ->  dst [cols][rows] bf16
__global__ void k_transpose(const float* __restrict__ src, unsigned short* __restrict__ dst,
                            int rows, int cols) {
  int idx = blockIdx.x * blockDim.x + threadIdx.x;
  if (idx >= rows * cols) return;
  int k = idx / cols, n = idx - k * cols;
  dst[(size_t)n * rows + k] = f2bf(src[idx]);
}

// ---- QKV projection GEMM ----------------------------------------------------
// 128x128 block tile, 8 waves, each wave 32x64 (2x4 WMMA accums).
// Branch-free async double-buffered pipeline (peeled tail). Epilogue scatters
// into Qh/Kh (head-major, Q pre-scaled by 1/8) and transposed Vt ([b][h][d][n]).

__global__ void __launch_bounds__(256)
k_gemm_qkv(const unsigned short* __restrict__ Xb, const unsigned short* __restrict__ WT,
           unsigned short* __restrict__ Qh, unsigned short* __restrict__ Kh,
           unsigned short* __restrict__ Vt) {
  __shared__ unsigned short As[2][128 * 32];  // [m][k]
  __shared__ unsigned short Bs[2][128 * 32];  // [n][k]
  const int t = threadIdx.x;
  const int wave = t >> 5, lane = t & 31;
  const int half = lane >> 4, ln = lane & 15;
  const int wm = (wave & 3) * 32, wn = (wave >> 2) * 64;
  const int row0 = blockIdx.y * 128, col0 = blockIdx.x * 128;
  f32x8 acc[2][4] = {};

  // Per-thread staging: both b128s land in the same row, adjacent 16B segments.
  const int srow = t >> 1, scol = (t & 1) * 16;
  const unsigned short* gA = Xb + (size_t)(row0 + srow) * DD + scol;
  const unsigned short* gB = WT + (size_t)(col0 + srow) * DD + scol;
  const unsigned loffA[2] = { lds_off(&As[0][srow * 32 + scol]),
                              lds_off(&As[1][srow * 32 + scol]) };
  const unsigned loffB[2] = { lds_off(&Bs[0][srow * 32 + scol]),
                              lds_off(&Bs[1][srow * 32 + scol]) };

  auto issue = [&](int buf) {
    async_ld_b128(loffA[buf],      gA);
    async_ld_b128(loffA[buf] + 16, gA + 8);
    async_ld_b128(loffB[buf],      gB);
    async_ld_b128(loffB[buf] + 16, gB + 8);
    gA += 32; gB += 32;
  };
  auto compute = [&](int buf) {
    const unsigned short* A  = &As[buf][0];
    const unsigned short* Bp = &Bs[buf][0];
    bf16x16 a0 = load_A_frag(A, wm + ln,      32, 0, half);
    bf16x16 a1 = load_A_frag(A, wm + 16 + ln, 32, 0, half);
#pragma unroll
    for (int j = 0; j < 4; ++j) {
      bf16x16 b = load_B_frag(Bp, wn + j * 16 + ln, 32, 0, half);
      acc[0][j] = wmma_bf16(a0, b, acc[0][j]);
      acc[1][j] = wmma_bf16(a1, b, acc[1][j]);
    }
  };

  issue(0);                                    // tile 0 -> buf0
  for (int kk = 0; kk < DD - 64; kk += 64) {   // steady state: no branches
    wait_async0(); __syncthreads(); issue(1); compute(0);
    wait_async0(); __syncthreads(); issue(0); compute(1);
  }
  wait_async0(); __syncthreads(); issue(1); compute(0);   // tile DD-64
  wait_async0(); __syncthreads();           compute(1);   // tile DD-32

#pragma unroll
  for (int mi = 0; mi < 2; ++mi)
#pragma unroll
    for (int j = 0; j < 4; ++j) {
      int c = col0 + wn + j * 16 + ln;
      int s = c >> 10, h = (c >> 6) & 15, d = c & 63;
#pragma unroll
      for (int r = 0; r < 8; ++r) {
        int grow = row0 + wm + mi * 16 + r + 8 * half;
        int b = grow >> 11, nidx = grow & (NN - 1);
        float v = acc[mi][j][r];
        if (s == 0)
          Qh[((size_t)(b * HH + h) * NN + nidx) * HD + d] = f2bf(v * 0.125f);
        else if (s == 1)
          Kh[((size_t)(b * HH + h) * NN + nidx) * HD + d] = f2bf(v);
        else
          Vt[((size_t)(b * HH + h) * HD + d) * NN + nidx] = f2bf(v);
      }
    }
}

// ---- flash attention --------------------------------------------------------
// One wave owns 16 queries; block = 8 waves = 128 queries of one (b,h).
// K/V chunks async double-buffered (branch-free pipeline, peeled tail);
// online softmax; P transposed via per-wave LDS staging.

__global__ void __launch_bounds__(256)
k_attn(const unsigned short* __restrict__ Qh, const unsigned short* __restrict__ Kh,
       const unsigned short* __restrict__ Vt, unsigned short* __restrict__ attn) {
  __shared__ unsigned short Ks[2][32 * 64];     // [key][d]
  __shared__ unsigned short Vs[2][64 * 32];     // [d][key]   (transposed V)
  __shared__ unsigned short Ps[8][16 * 32];     // per-wave P staging [q][key]
  const int t = threadIdx.x, wave = t >> 5, lane = t & 31;
  const int half = lane >> 4, ln = lane & 15;
  const int ntiles = NN / 128;
  const int bh = blockIdx.x / ntiles, qt = blockIdx.x % ntiles;
  const int b = bh / HH, h = bh % HH;
  const unsigned short* Qb = Qh + (size_t)bh * NN * HD;
  const int q0 = qt * 128 + wave * 16;

  bf16x16 qf[2];
#pragma unroll
  for (int f = 0; f < 2; ++f)
    qf[f] = load_A_frag(Qb, q0 + ln, HD, f * 32, half);

  f32x8 accO[4] = {};
  float mi[8], li[8];
#pragma unroll
  for (int r = 0; r < 8; ++r) { mi[r] = -1e30f; li[r] = 0.f; }

  // Per-thread staging addresses (advance per 32-key chunk).
  const int kkey = t >> 3, ksg = t & 7;          // K: 32 rows * 8 uint4
  const int vd = t >> 2, vsg = t & 3;            // V: 64 rows * 4 uint4
  const unsigned short* gK = Kh + (size_t)bh * NN * HD + (size_t)kkey * HD + ksg * 8;
  const unsigned short* gV = Vt + (size_t)bh * HD * NN + (size_t)vd * NN + vsg * 8;
  const unsigned loffK[2] = { lds_off(&Ks[0][kkey * 64 + ksg * 8]),
                              lds_off(&Ks[1][kkey * 64 + ksg * 8]) };
  const unsigned loffV[2] = { lds_off(&Vs[0][vd * 32 + vsg * 8]),
                              lds_off(&Vs[1][vd * 32 + vsg * 8]) };

  auto issue = [&](int buf) {
    async_ld_b128(loffK[buf], gK);
    async_ld_b128(loffV[buf], gV);
    gK += 32 * HD;   // next 32 keys
    gV += 32;
  };
  auto compute = [&](int buf) {
    // S = Q * K^T  (16 x 32 keys), contraction over d=64
    f32x8 s0 = {}, s1 = {};
#pragma unroll
    for (int f = 0; f < 2; ++f) {
      s0 = wmma_bf16(qf[f], load_B_frag(&Ks[buf][0], ln,      HD, f * 32, half), s0);
      s1 = wmma_bf16(qf[f], load_B_frag(&Ks[buf][0], 16 + ln, HD, f * 32, half), s1);
    }
    // online softmax (rows r+8*half; stats reduced across lanes 0..15 per half)
    float alpha[8];
#pragma unroll
    for (int r = 0; r < 8; ++r) {
      float v = fmaxf(s0[r], s1[r]);
      v = fmaxf(v, __shfl_xor(v, 1));
      v = fmaxf(v, __shfl_xor(v, 2));
      v = fmaxf(v, __shfl_xor(v, 4));
      v = fmaxf(v, __shfl_xor(v, 8));
      float mn = fmaxf(mi[r], v);
      alpha[r] = __expf(mi[r] - mn);
      mi[r] = mn;
      float p0 = __expf(s0[r] - mn);
      float p1 = __expf(s1[r] - mn);
      float rs = p0 + p1;
      rs += __shfl_xor(rs, 1);
      rs += __shfl_xor(rs, 2);
      rs += __shfl_xor(rs, 4);
      rs += __shfl_xor(rs, 8);
      li[r] = li[r] * alpha[r] + rs;
      Ps[wave][(r + 8 * half) * 32 + ln]      = f2bf(p0);
      Ps[wave][(r + 8 * half) * 32 + 16 + ln] = f2bf(p1);
    }
#pragma unroll
    for (int j = 0; j < 4; ++j)
#pragma unroll
      for (int r = 0; r < 8; ++r) accO[j][r] *= alpha[r];
    // same-wave LDS RAW: drain DS before re-reading P in transposed layout
    asm volatile("s_wait_dscnt 0" ::: "memory");
    bf16x16 pa = load_A_frag(Ps[wave], ln, 32, 0, half);
#pragma unroll
    for (int j = 0; j < 4; ++j)
      accO[j] = wmma_bf16(pa, load_B_frag(&Vs[buf][0], j * 16 + ln, 32, 0, half),
                          accO[j]);
  };

  issue(0);                                      // chunk 0 -> buf0
  for (int c = 0; c < NN / 32 - 2; c += 2) {     // 31 branch-free iterations
    wait_async0(); __syncthreads(); issue(1); compute(0);
    wait_async0(); __syncthreads(); issue(0); compute(1);
  }
  wait_async0(); __syncthreads(); issue(1); compute(0);   // chunk NC-2
  wait_async0(); __syncthreads();           compute(1);   // chunk NC-1

#pragma unroll
  for (int j = 0; j < 4; ++j)
#pragma unroll
    for (int r = 0; r < 8; ++r) {
      int q = q0 + r + 8 * half;
      float v = accO[j][r] / li[r];
      attn[((size_t)(b * NN + q)) * DD + h * HD + j * 16 + ln] = f2bf(v);
    }
}

// ---- output projection GEMM (+bias, f32 out) --------------------------------

__global__ void __launch_bounds__(256)
k_gemm_proj(const unsigned short* __restrict__ Ab, const unsigned short* __restrict__ WT,
            const float* __restrict__ bias, float* __restrict__ out) {
  __shared__ unsigned short As[2][128 * 32];
  __shared__ unsigned short Bs[2][128 * 32];
  const int t = threadIdx.x;
  const int wave = t >> 5, lane = t & 31;
  const int half = lane >> 4, ln = lane & 15;
  const int wm = (wave & 3) * 32, wn = (wave >> 2) * 64;
  const int row0 = blockIdx.y * 128, col0 = blockIdx.x * 128;
  f32x8 acc[2][4] = {};

  const int srow = t >> 1, scol = (t & 1) * 16;
  const unsigned short* gA = Ab + (size_t)(row0 + srow) * DD + scol;
  const unsigned short* gB = WT + (size_t)(col0 + srow) * DD + scol;
  const unsigned loffA[2] = { lds_off(&As[0][srow * 32 + scol]),
                              lds_off(&As[1][srow * 32 + scol]) };
  const unsigned loffB[2] = { lds_off(&Bs[0][srow * 32 + scol]),
                              lds_off(&Bs[1][srow * 32 + scol]) };

  auto issue = [&](int buf) {
    async_ld_b128(loffA[buf],      gA);
    async_ld_b128(loffA[buf] + 16, gA + 8);
    async_ld_b128(loffB[buf],      gB);
    async_ld_b128(loffB[buf] + 16, gB + 8);
    gA += 32; gB += 32;
  };
  auto compute = [&](int buf) {
    const unsigned short* A  = &As[buf][0];
    const unsigned short* Bp = &Bs[buf][0];
    bf16x16 a0 = load_A_frag(A, wm + ln,      32, 0, half);
    bf16x16 a1 = load_A_frag(A, wm + 16 + ln, 32, 0, half);
#pragma unroll
    for (int j = 0; j < 4; ++j) {
      bf16x16 b = load_B_frag(Bp, wn + j * 16 + ln, 32, 0, half);
      acc[0][j] = wmma_bf16(a0, b, acc[0][j]);
      acc[1][j] = wmma_bf16(a1, b, acc[1][j]);
    }
  };

  issue(0);
  for (int kk = 0; kk < DD - 64; kk += 64) {
    wait_async0(); __syncthreads(); issue(1); compute(0);
    wait_async0(); __syncthreads(); issue(0); compute(1);
  }
  wait_async0(); __syncthreads(); issue(1); compute(0);
  wait_async0(); __syncthreads();           compute(1);

#pragma unroll
  for (int mi = 0; mi < 2; ++mi)
#pragma unroll
    for (int j = 0; j < 4; ++j) {
      int c = col0 + wn + j * 16 + ln;
      float bv = bias[c];
#pragma unroll
      for (int r = 0; r < 8; ++r) {
        int grow = row0 + wm + mi * 16 + r + 8 * half;
        out[(size_t)grow * DD + c] = acc[mi][j][r] + bv;
      }
    }
}

// ---- host launcher ----------------------------------------------------------

extern "C" void kernel_launch(void* const* d_in, const int* in_sizes, int n_in,
                              void* d_out, int out_size, void* d_ws, size_t ws_size,
                              hipStream_t stream) {
  const float* X     = (const float*)d_in[0];   // [B,N,D]
  const float* Wqkv  = (const float*)d_in[1];   // [D,3D]
  const float* Wproj = (const float*)d_in[2];   // [D,D]
  const float* bproj = (const float*)d_in[3];   // [D]

  char* ws = (char*)d_ws;
  size_t off = 0;
  unsigned short* Xb     = (unsigned short*)(ws + off); off += (size_t)M * DD * 2;
  unsigned short* WqkvT  = (unsigned short*)(ws + off); off += (size_t)NOUT * DD * 2;
  unsigned short* WprojT = (unsigned short*)(ws + off); off += (size_t)DD * DD * 2;
  unsigned short* Qh     = (unsigned short*)(ws + off); off += (size_t)BB * HH * NN * HD * 2;
  unsigned short* Kh     = (unsigned short*)(ws + off); off += (size_t)BB * HH * NN * HD * 2;
  unsigned short* Vt     = (unsigned short*)(ws + off); off += (size_t)BB * HH * NN * HD * 2;
  unsigned short* attnb  = (unsigned short*)(ws + off); off += (size_t)M * DD * 2;

  int n8 = M * DD / 8;
  k_convert<<<(n8 + 255) / 256, 256, 0, stream>>>(X, Xb, n8);
  k_transpose<<<(DD * NOUT + 255) / 256, 256, 0, stream>>>(Wqkv, WqkvT, DD, NOUT);
  k_transpose<<<(DD * DD + 255) / 256, 256, 0, stream>>>(Wproj, WprojT, DD, DD);

  k_gemm_qkv<<<dim3(NOUT / 128, M / 128), 256, 0, stream>>>(Xb, WqkvT, Qh, Kh, Vt);
  k_attn<<<BB * HH * (NN / 128), 256, 0, stream>>>(Qh, Kh, Vt, attnb);
  k_gemm_proj<<<dim3(DD / 128, M / 128), 256, 0, stream>>>(attnb, WprojT, bproj,
                                                           (float*)d_out);
}
